// PointSampler_38190849196537
// MI455X (gfx1250) — matrix operand order repaired and checked by reading, not compiled, not run on hardware
//
#include <hip/hip_runtime.h>
#include <hip/hip_bf16.h>

typedef __attribute__((ext_vector_type(16))) _Float16 v16h;
typedef __attribute__((ext_vector_type(8)))  float    v8f;

#define NPTS 50000
#define NEDG 800000
#define MPTS 25000
#define KNB  15
#define HDIM 128
#define BN_EPS 1e-5f

// ---------------------------------------------------------------- utilities
__global__ void zero_f32(float* __restrict__ p, int n) {
    int i = blockIdx.x * blockDim.x + threadIdx.x;
    if (i < n) p[i] = 0.0f;
}

// ------------------------------------------------- layer 1: [N,3]@[3,128]+b
__global__ void lin1_kernel(const float* __restrict__ pts,
                            const float* __restrict__ W1,
                            const float* __restrict__ b1,
                            float* __restrict__ out, int nrows) {
    int i = blockIdx.x * blockDim.x + threadIdx.x;
    if (i >= nrows * HDIM) return;
    int r = i >> 7, c = i & 127;
    float x = pts[r * 3 + 0], y = pts[r * 3 + 1], z = pts[r * 3 + 2];
    out[i] = x * W1[c] + y * W1[HDIM + c] + z * W1[2 * HDIM + c] + b1[c];
}

// --------------------------------------------- column sums / sumsq (atomic)
#define STAT_ROWS 256
__global__ void colstats_kernel(const float* __restrict__ X, int nrows,
                                float* __restrict__ sum, float* __restrict__ sq) {
    int c  = threadIdx.x;                 // 128 threads
    int r0 = blockIdx.x * STAT_ROWS;
    int r1 = r0 + STAT_ROWS; if (r1 > nrows) r1 = nrows;
    float s = 0.0f, q = 0.0f;
    for (int r = r0; r < r1; ++r) {
        float v = X[(size_t)r * HDIM + c];
        s += v; q += v * v;
    }
    atomicAdd(&sum[c], s);
    atomicAdd(&sq[c], q);
}

// -------------------------------------------------- BN (train mode) + ReLU
__global__ void bn_relu_kernel(float* __restrict__ X, int nrows,
                               const float* __restrict__ sum,
                               const float* __restrict__ sq,
                               const float* __restrict__ g,
                               const float* __restrict__ beta) {
    int i = blockIdx.x * blockDim.x + threadIdx.x;
    if (i >= nrows * HDIM) return;
    int c = i & 127;
    float invn = 1.0f / (float)nrows;
    float m = sum[c] * invn;
    float v = sq[c] * invn - m * m;       // biased variance (jnp.var default)
    float sc = g[c] * rsqrtf(v + BN_EPS);
    float y = sc * (X[i] - m) + beta[c];
    X[i] = y > 0.0f ? y : 0.0f;
}

// -------------------------------------------- WMMA GEMM: [nr,128]@[128,nc]+b
// One wave32 per 32-row super-strip (2 x 16-row WMMA strips sharing B frags).
// K=128 as 4 k-tiles of 32, f16 inputs / f32 accumulate.
// OOB rows/cols are CLAMPED (not zeroed): a garbage A-row only corrupts the
// matching unstored D-row; a garbage B-col only corrupts the matching
// unstored D-col. All loads are unconditional -> no exec-mask churn.

__device__ __forceinline__ v16h load_afrag(const float* __restrict__ Arow,
                                           int kt, int h) {
    // 16-bit A 16x32 layout: element e<8 -> k = kt*32 + 8h + e
    //                        element e>=8 -> k = kt*32 + 16 + 8h + (e-8)
    const float4* plo = (const float4*)(Arow + kt * 32 + 8 * h);
    const float4* phi = (const float4*)(Arow + kt * 32 + 16 + 8 * h);
    float4 lo0 = plo[0], lo1 = plo[1];
    float4 hi0 = phi[0], hi1 = phi[1];
    v16h a;
    a[0] = (_Float16)lo0.x;  a[1] = (_Float16)lo0.y;
    a[2] = (_Float16)lo0.z;  a[3] = (_Float16)lo0.w;
    a[4] = (_Float16)lo1.x;  a[5] = (_Float16)lo1.y;
    a[6] = (_Float16)lo1.z;  a[7] = (_Float16)lo1.w;
    a[8] = (_Float16)hi0.x;  a[9] = (_Float16)hi0.y;
    a[10] = (_Float16)hi0.z; a[11] = (_Float16)hi0.w;
    a[12] = (_Float16)hi1.x; a[13] = (_Float16)hi1.y;
    a[14] = (_Float16)hi1.z; a[15] = (_Float16)hi1.w;
    return a;
}

template<int NCOLS, int LDW>
__global__ void gemm128_wmma(const float* __restrict__ A,   // [nrows,128]
                             const float* __restrict__ W,   // [>=128, LDW]
                             const float* __restrict__ bias,// [NCOLS]
                             float* __restrict__ Y,         // [nrows,NCOLS]
                             int nrows) {
    const int wave = blockIdx.x * (blockDim.x >> 5) + (threadIdx.x >> 5);
    const int lane = threadIdx.x & 31;
    const int row0 = wave * 32;
    if (row0 >= nrows) return;            // wave-uniform
    const int l16 = lane & 15;
    const int h   = lane >> 4;

    // clamped A rows for the two strips (garbage rows are never stored)
    int arow0 = row0 + l16;       if (arow0 >= nrows) arow0 = nrows - 1;
    int arow1 = row0 + 16 + l16;  if (arow1 >= nrows) arow1 = nrows - 1;
    const float* Arow0 = A + (size_t)arow0 * HDIM;
    const float* Arow1 = A + (size_t)arow1 * HDIM;

    v16h af0[4], af1[4];
#pragma unroll
    for (int kt = 0; kt < 4; ++kt) {
        af0[kt] = load_afrag(Arow0, kt, h);
        af1[kt] = load_afrag(Arow1, kt, h);
    }

    const int NT = (NCOLS + 15) >> 4;
    for (int nt = 0; nt < NT; ++nt) {
        const int col  = nt * 16 + l16;
        int colc = col;
        if (NCOLS & 15) { if (colc >= NCOLS) colc = NCOLS - 1; }   // clamp
        // all 16 B elements are immediate offsets from this per-lane base
        const float* Wp = W + colc + 8 * h * LDW;
        v8f acc0 = {}, acc1 = {};
#pragma unroll
        for (int kt = 0; kt < 4; ++kt) {
            v16h b;
#pragma unroll
            for (int e = 0; e < 16; ++e)     // B 32x16: k = kt*32 + 8h + e (e<8), kt*32 + 16 + 8h + (e-8) (e>=8)
                b[e] = (_Float16)Wp[(kt * 32 + e + (e >= 8 ? 8 : 0)) * LDW];
            acc0 = __builtin_amdgcn_wmma_f32_16x16x32_f16(
                false, af0[kt], false, b, (short)0, acc0, false, false);
            acc1 = __builtin_amdgcn_wmma_f32_16x16x32_f16(
                false, af1[kt], false, b, (short)0, acc1, false, false);
        }
        const bool colok = ((NCOLS & 15) == 0) || (col < NCOLS);
        if (colok) {
            float bv = bias[col];
            float* Ycol = Y + col;
            if (row0 + 32 <= nrows) {        // uniform fast path
#pragma unroll
                for (int r = 0; r < 8; ++r) {
                    Ycol[(size_t)(row0 + r + 8 * h) * NCOLS]      = acc0[r] + bv;
                    Ycol[(size_t)(row0 + 16 + r + 8 * h) * NCOLS] = acc1[r] + bv;
                }
            } else {                         // ragged last super-strip
#pragma unroll
                for (int r = 0; r < 8; ++r) {
                    int r0i = row0 + r + 8 * h;
                    int r1i = row0 + 16 + r + 8 * h;
                    if (r0i < nrows) Ycol[(size_t)r0i * NCOLS] = acc0[r] + bv;
                    if (r1i < nrows) Ycol[(size_t)r1i * NCOLS] = acc1[r] + bv;
                }
            }
        }
    }
}

// ---------------------------------------------------- GNN edge scatter-add
__global__ void gnn_scatter_kernel(const float* __restrict__ msg, // xWn [N,128]
                                   const int* __restrict__ src,
                                   const int* __restrict__ dst,
                                   float* __restrict__ agg) {
    int t = blockIdx.x * blockDim.x + threadIdx.x;   // 32 threads per edge
    int e = t >> 5;
    if (e >= NEDG) return;
    int c4 = (t & 31) * 4;
    const float* ms = msg + (size_t)src[e] * HDIM + c4;
    float*       ag = agg + (size_t)dst[e] * HDIM + c4;
    atomicAdd(&ag[0], ms[0]);
    atomicAdd(&ag[1], ms[1]);
    atomicAdd(&ag[2], ms[2]);
    atomicAdd(&ag[3], ms[3]);
}

__global__ void gnn_count_kernel(const int* __restrict__ dst, float* __restrict__ cnt) {
    int e = blockIdx.x * blockDim.x + threadIdx.x;
    if (e < NEDG) atomicAdd(&cnt[dst[e]], 1.0f);
}

// ----------------------------------- x3 = relu(xWc + agg / max(cnt,1))
__global__ void gnn_combine_kernel(float* __restrict__ xWc,        // in/out
                                   const float* __restrict__ agg,
                                   const float* __restrict__ cnt, int nrows) {
    int i = blockIdx.x * blockDim.x + threadIdx.x;
    if (i >= nrows * HDIM) return;
    int r = i >> 7;
    float inv = 1.0f / fmaxf(cnt[r], 1.0f);
    float y = xWc[i] + agg[i] * inv;
    xWc[i] = y > 0.0f ? y : 0.0f;
}

// ---------------------------------------------------- gather sampled points
__global__ void gather_kernel(const float* __restrict__ x3,
                              const float* __restrict__ pts,
                              const int* __restrict__ sidx,
                              float* __restrict__ sx,   // [M,128] -> d_out tail
                              float* __restrict__ sp) { // [M,3]
    int i = blockIdx.x * blockDim.x + threadIdx.x;
    if (i >= MPTS * HDIM) return;
    int m = i >> 7, c = i & 127;
    int id = sidx[m];
    sx[i] = x3[(size_t)id * HDIM + c];
    if (c < 3) sp[m * 3 + c] = pts[id * 3 + c];
}

// ------------------------------------------------ attention: 1 wave / point
__global__ void attention_kernel(const float* __restrict__ q,     // [M,64]
                                 const float* __restrict__ xk,    // [M,64]
                                 const float* __restrict__ P,     // [M,131] sx@Wcat_top+bcat
                                 const float* __restrict__ sp,    // [M,3]
                                 const int* __restrict__ nbr,     // [M,15]
                                 const float* __restrict__ Wcat,  // [131,131]
                                 const float* __restrict__ Wout,  // [131,3]
                                 const float* __restrict__ bout,  // [3]
                                 float* __restrict__ refined) {   // [M,3]
    const int m = blockIdx.x * (blockDim.x >> 5) + (threadIdx.x >> 5);
    const int lane = threadIdx.x & 31;
    if (m >= MPTS) return;                 // wave-uniform

    const float spx = sp[m * 3 + 0], spy = sp[m * 3 + 1], spz = sp[m * 3 + 2];

    // lane j (<15): score_j = (q_m . xk[nb_j]) / scale * mask_j  (mask->0 quirk)
    float score = -INFINITY;
    int   nb = 0;
    if (lane < KNB) {
        nb = nbr[m * KNB + lane];
        const float* qv = q  + (size_t)m  * 64;
        const float* kv = xk + (size_t)nb * 64;
        float s = 0.0f;
#pragma unroll 8
        for (int d = 0; d < 64; ++d) s += qv[d] * kv[d];
        float maskv = (nb != 0) ? 1.0f : 0.0f;
        score = (s / (8.0f + 1e-6f)) * maskv;
    }
    // softmax over 15 lanes (lanes>=15 contribute -inf / 0)
    float mx = score;
#pragma unroll
    for (int o = 16; o >= 1; o >>= 1) mx = fmaxf(mx, __shfl_xor(mx, o, 32));
    float ex = (lane < KNB) ? __expf(score - mx) : 0.0f;
    float den = ex;
#pragma unroll
    for (int o = 16; o >= 1; o >>= 1) den += __shfl_xor(den, o, 32);
    float attn = ex / den;

    // broadcast attn_j and nb_j to all lanes
    float attnv[KNB]; int nbj[KNB];
#pragma unroll
    for (int j = 0; j < KNB; ++j) {
        attnv[j] = __shfl(attn, j, 32);
        nbj[j]   = __shfl(nb,   j, 32);
    }
    // weighted relative position: wr = sum_j attn_j * (sp[nb_j] - sp[m])
    float wrx = 0.0f, wry = 0.0f, wrz = 0.0f;
    if (lane < KNB) {
        wrx = attn * (sp[nb * 3 + 0] - spx);
        wry = attn * (sp[nb * 3 + 1] - spy);
        wrz = attn * (sp[nb * 3 + 2] - spz);
    }
#pragma unroll
    for (int o = 16; o >= 1; o >>= 1) {
        wrx += __shfl_xor(wrx, o, 32);
        wry += __shfl_xor(wry, o, 32);
        wrz += __shfl_xor(wrz, o, 32);
    }
    // aggf[d] = sum_j attn_j * P[nb_j][d] + wr . Wcat[128..130][d]
    float d0 = 0.0f, d1 = 0.0f, d2 = 0.0f;
    for (int d = lane; d < 131; d += 32) {
        float a = wrx * Wcat[128 * 131 + d] + wry * Wcat[129 * 131 + d]
                + wrz * Wcat[130 * 131 + d];
#pragma unroll
        for (int j = 0; j < KNB; ++j) a += attnv[j] * P[(size_t)nbj[j] * 131 + d];
        d0 += a * Wout[d * 3 + 0];
        d1 += a * Wout[d * 3 + 1];
        d2 += a * Wout[d * 3 + 2];
    }
#pragma unroll
    for (int o = 16; o >= 1; o >>= 1) {
        d0 += __shfl_xor(d0, o, 32);
        d1 += __shfl_xor(d1, o, 32);
        d2 += __shfl_xor(d2, o, 32);
    }
    if (lane == 0) {
        refined[m * 3 + 0] = spx + d0 + bout[0];
        refined[m * 3 + 1] = spy + d1 + bout[1];
        refined[m * 3 + 2] = spz + d2 + bout[2];
    }
}

// ---------------------------------------------------------------- launcher
extern "C" void kernel_launch(void* const* d_in, const int* in_sizes, int n_in,
                              void* d_out, int out_size, void* d_ws, size_t ws_size,
                              hipStream_t stream) {
    const float* points = (const float*)d_in[0];
    const int*   eidx   = (const int*)d_in[1];
    const int*   sidx   = (const int*)d_in[2];
    const int*   nbr    = (const int*)d_in[3];
    const float* W1 = (const float*)d_in[4],  *b1 = (const float*)d_in[5];
    const float* g1 = (const float*)d_in[6],  *be1= (const float*)d_in[7];
    const float* W2 = (const float*)d_in[8],  *b2 = (const float*)d_in[9];
    const float* g2 = (const float*)d_in[10], *be2= (const float*)d_in[11];
    const float* Wc = (const float*)d_in[12], *bc = (const float*)d_in[13];
    const float* Wn = (const float*)d_in[14], *bnb= (const float*)d_in[15];
    const float* Wq = (const float*)d_in[16], *bq = (const float*)d_in[17];
    const float* Wk = (const float*)d_in[18], *bk = (const float*)d_in[19];
    const float* Wcat=(const float*)d_in[20], *bcat=(const float*)d_in[21];
    const float* Wout=(const float*)d_in[22], *bout=(const float*)d_in[23];

    const int* srcE = eidx;            // edge_index[0]
    const int* dstE = eidx + NEDG;     // edge_index[1]

    float* refined = (float*)d_out;                 // [M,3]
    float* sx      = (float*)d_out + MPTS * 3;      // [M,128]

    // workspace layout (floats)
    float* ws = (float*)d_ws;
    size_t off = 0;
    float* bufA = ws + off; off += (size_t)NPTS * HDIM;   // t1/x1 -> xWn
    float* bufB = ws + off; off += (size_t)NPTS * HDIM;   // t2/x2 -> agg
    float* bufC = ws + off; off += (size_t)NPTS * HDIM;   // xWc -> x3
    float* cnt  = ws + off; off += NPTS;
    float* stat = ws + off; off += 256;                   // sum[128] | sumsq[128]
    float* sp   = ws + off; off += (size_t)MPTS * 3;
    float* qb   = ws + off; off += (size_t)MPTS * 64;
    float* kb   = ws + off; off += (size_t)MPTS * 64;
    float* Pb   = ws + off; off += (size_t)MPTS * 131;
    (void)ws_size; (void)in_sizes; (void)n_in; (void)out_size;

    const int NH = NPTS * HDIM;
    const int statBlocks = (NPTS + STAT_ROWS - 1) / STAT_ROWS;
    const int gemmBlocksN = ((NPTS + 31) / 32 + 7) / 8;   // 8 waves / 256-thr block
    const int gemmBlocksM = ((MPTS + 31) / 32 + 7) / 8;

    // ---- layer 1: linear + BN + ReLU
    lin1_kernel<<<(NH + 255) / 256, 256, 0, stream>>>(points, W1, b1, bufA, NPTS);
    zero_f32<<<1, 256, 0, stream>>>(stat, 256);
    colstats_kernel<<<statBlocks, 128, 0, stream>>>(bufA, NPTS, stat, stat + 128);
    bn_relu_kernel<<<(NH + 255) / 256, 256, 0, stream>>>(bufA, NPTS, stat, stat + 128, g1, be1);

    // ---- layer 2: WMMA GEMM + BN + ReLU
    gemm128_wmma<HDIM, HDIM><<<gemmBlocksN, 256, 0, stream>>>(bufA, W2, b2, bufB, NPTS);
    zero_f32<<<1, 256, 0, stream>>>(stat, 256);
    colstats_kernel<<<statBlocks, 128, 0, stream>>>(bufB, NPTS, stat, stat + 128);
    bn_relu_kernel<<<(NH + 255) / 256, 256, 0, stream>>>(bufB, NPTS, stat, stat + 128, g2, be2);

    // ---- GNN: xWn (A), xWc (C), scatter-mean into B, combine into C
    gemm128_wmma<HDIM, HDIM><<<gemmBlocksN, 256, 0, stream>>>(bufB, Wn, bnb, bufA, NPTS);
    gemm128_wmma<HDIM, HDIM><<<gemmBlocksN, 256, 0, stream>>>(bufB, Wc, bc, bufC, NPTS);
    zero_f32<<<(NH + 255) / 256, 256, 0, stream>>>(bufB, NH);
    zero_f32<<<(NPTS + 255) / 256, 256, 0, stream>>>(cnt, NPTS);
    gnn_scatter_kernel<<<(NEDG * 32 + 255) / 256, 256, 0, stream>>>(bufA, srcE, dstE, bufB);
    gnn_count_kernel<<<(NEDG + 255) / 256, 256, 0, stream>>>(dstE, cnt);
    gnn_combine_kernel<<<(NH + 255) / 256, 256, 0, stream>>>(bufC, bufB, cnt, NPTS);

    // ---- sampling gather (sx straight into d_out)
    gather_kernel<<<(MPTS * HDIM + 255) / 256, 256, 0, stream>>>(bufC, points, sidx, sx, sp);

    // ---- attention projections (gather commutes with linear maps)
    gemm128_wmma<64, 64><<<gemmBlocksM, 256, 0, stream>>>(sx, Wq, bq, qb, MPTS);
    gemm128_wmma<64, 64><<<gemmBlocksM, 256, 0, stream>>>(sx, Wk, bk, kb, MPTS);
    gemm128_wmma<131, 131><<<gemmBlocksM, 256, 0, stream>>>(sx, Wcat, bcat, Pb, MPTS);

    // ---- attention + output projection + residual
    attention_kernel<<<(MPTS + 3) / 4, 128, 0, stream>>>(qb, kb, Pb, sp, nbr,
                                                         Wcat, Wout, bout, refined);
}
